// CausalSelfAttention_31722628448860
// MI455X (gfx1250) — matrix operand order repaired and checked by reading, not compiled, not run on hardware
//
#include <hip/hip_runtime.h>
#include <stdint.h>

typedef __attribute__((ext_vector_type(16))) __bf16 v16bf;
typedef __attribute__((ext_vector_type(8)))  float  v8f;

#define DMODEL 1024
#define NHEADS 16
#define HDIM   64
#define SEQ    2048
#define BATCH  2
#define MROWS  (BATCH * SEQ)   // 4096

union BF16Frag { v16bf v; unsigned u[8]; };
union F8       { v8f   v; float    f[8]; };

static __device__ __forceinline__ float red_max16(float x) {
    x = fmaxf(x, __shfl_xor(x, 1, 32));
    x = fmaxf(x, __shfl_xor(x, 2, 32));
    x = fmaxf(x, __shfl_xor(x, 4, 32));
    x = fmaxf(x, __shfl_xor(x, 8, 32));
    return x;
}
static __device__ __forceinline__ float red_sum16(float x) {
    x += __shfl_xor(x, 1, 32);
    x += __shfl_xor(x, 2, 32);
    x += __shfl_xor(x, 4, 32);
    x += __shfl_xor(x, 8, 32);
    return x;
}

// async global -> LDS, 16 bytes per lane, tracked by ASYNCcnt
static __device__ __forceinline__ void async_b128(unsigned lds_off, const void* g) {
    asm volatile("global_load_async_to_lds_b128 %0, %1, off"
                 :: "v"(lds_off), "v"(g) : "memory");
}

// ---------------------------------------------------------------- converts
__global__ void f32_to_bf16_kernel(const float* __restrict__ src,
                                   __bf16* __restrict__ dst, int n) {
    int i = blockIdx.x * blockDim.x + threadIdx.x;
    if (i < n) dst[i] = (__bf16)src[i];
}

// dst[n*K + k] = src[k*N + n]  (1024x1024 weight transpose + bf16 convert)
__global__ void f32_to_bf16_T_kernel(const float* __restrict__ src,
                                     __bf16* __restrict__ dst) {
    int i = blockIdx.x * blockDim.x + threadIdx.x;   // over 1024*1024
    int n = i >> 10;
    int k = i & 1023;
    dst[i] = (__bf16)src[k * DMODEL + n];
}

// ---------------------------------------------------------------- GEMM
// C[4096,1024] = A[4096,1024](bf16) @ W[1024,1024] + bias, with W given
// TRANSPOSED in memory: Wt[n][k].  LDS tiles are straight async copies.
// mode 0: f32 row-major out        (final projection)
// mode 1: bf16 out [B,H,S,HDIM]    (Q,K projections)
// mode 2: bf16 out [B,H,HDIM,S]    (V projection, pre-transposed for flash)
__global__ __launch_bounds__(128)
void gemm_bf16_kernel(const __bf16* __restrict__ A, const __bf16* __restrict__ Wt,
                      const float* __restrict__ bias,
                      float* __restrict__ outF, __bf16* __restrict__ outH,
                      int mode) {
    __shared__ __bf16 As[2][64][40];   // [m][k] tile, row stride 80B (16B multiple)
    __shared__ __bf16 Bs[2][64][40];   // [n][k] tile

    const int tid  = threadIdx.x;
    const int lane = tid & 31;
    const int w    = tid >> 5;
    const int hl   = lane >> 4;
    const int l16  = lane & 15;
    const int tileN = blockIdx.x * 64;
    const int tileM = blockIdx.y * 64;

    const unsigned asBase = (unsigned)(uintptr_t)&As[0][0][0];
    const unsigned bsBase = (unsigned)(uintptr_t)&Bs[0][0][0];
    const unsigned BUFSZ  = 64 * 40 * 2;   // 5120 B per buffer

    F8 acc[4];
#pragma unroll
    for (int nt = 0; nt < 4; ++nt)
#pragma unroll
        for (int r = 0; r < 8; ++r) acc[nt].f[r] = 0.0f;

    // issue one 64x32 A tile + 64x32 B tile (4 async b128 ops per thread)
#define GEMM_ISSUE(k0, buf)                                                        \
    {                                                                              \
        _Pragma("unroll")                                                          \
        for (int i = 0; i < 2; ++i) {                                              \
            int q = tid + 128 * i; int m = q >> 2, c = q & 3;                      \
            async_b128(asBase + (buf) * BUFSZ + m * 80 + c * 16,                   \
                       A + (size_t)(tileM + m) * DMODEL + (k0) + 8 * c);           \
        }                                                                          \
        _Pragma("unroll")                                                          \
        for (int i = 0; i < 2; ++i) {                                              \
            int q = tid + 128 * i; int n = q >> 2, c = q & 3;                      \
            async_b128(bsBase + (buf) * BUFSZ + n * 80 + c * 16,                   \
                       Wt + (size_t)(tileN + n) * DMODEL + (k0) + 8 * c);          \
        }                                                                          \
    }

    GEMM_ISSUE(0, 0);

    for (int it = 0; it < DMODEL / 32; ++it) {
        const int buf = it & 1;
        if (it + 1 < DMODEL / 32) {
            GEMM_ISSUE((it + 1) * 32, buf ^ 1);
            asm volatile("s_wait_asynccnt 0x4" ::: "memory");   // tile `it` landed
        } else {
            asm volatile("s_wait_asynccnt 0x0" ::: "memory");
        }
        __syncthreads();

        BF16Frag af;
        {
            int m = 16 * w + l16;
#pragma unroll
            for (int i = 0; i < 8; ++i) {
                int k = (i < 4) ? (2 * i + 8 * hl) : (16 + 2 * (i - 4) + 8 * hl);
                af.u[i] = *(const unsigned*)&As[buf][m][k];
            }
        }
#pragma unroll
        for (int nt = 0; nt < 4; ++nt) {
            BF16Frag bf;
#pragma unroll
            for (int i = 0; i < 8; ++i) {
                int k = (i < 4) ? (2 * i + 8 * hl) : (16 + 2 * (i - 4) + 8 * hl);
                bf.u[i] = *(const unsigned*)&Bs[buf][nt * 16 + l16][k];
            }
            acc[nt].v = __builtin_amdgcn_wmma_f32_16x16x32_bf16(
                false, af.v, false, bf.v, (short)0, acc[nt].v, false, false);
        }
        __syncthreads();
    }
#undef GEMM_ISSUE

    // epilogue: C/D layout -> M = r + 8*(lane>=16), N = lane%16
#pragma unroll
    for (int nt = 0; nt < 4; ++nt)
#pragma unroll
        for (int r = 0; r < 8; ++r) {
            int mg  = tileM + 16 * w + r + 8 * hl;
            int col = tileN + nt * 16 + l16;
            float v = acc[nt].f[r] + bias[col];
            if (mode == 0) {
                outF[(size_t)mg * DMODEL + col] = v;
            } else {
                int b = mg >> 11;           // / SEQ
                int s = mg & (SEQ - 1);
                int h = col >> 6;           // / HDIM
                int d = col & (HDIM - 1);
                if (mode == 1)
                    outH[(((size_t)(b * NHEADS + h) * SEQ) + s) * HDIM + d] = (__bf16)v;
                else  // mode 2: V stored [B,H,HDIM,S]
                    outH[(((size_t)(b * NHEADS + h) * HDIM) + d) * SEQ + s] = (__bf16)v;
            }
        }
}

// ---------------------------------------------------------------- flash attention
// grid: (S/64, NHEADS, BATCH), 128 threads (4 waves)
// Qh,Kh: [B,H,S,HDIM] bf16 ; Vt: [B,H,HDIM,S] bf16 (pre-transposed)
__global__ __launch_bounds__(128)
void flash_attn_kernel(const __bf16* __restrict__ Qh, const __bf16* __restrict__ Kh,
                       const __bf16* __restrict__ Vt, __bf16* __restrict__ AO) {
    __shared__ __bf16 Kt [2][64][72];   // [t][d], row stride 144B (16B multiple)
    __shared__ __bf16 VtT[2][64][72];   // [d][t]
    __shared__ __bf16 Pt [64][72];      // [m][t], wave-private rows

    const int tid  = threadIdx.x;
    const int lane = tid & 31;
    const int w    = tid >> 5;
    const int hl   = lane >> 4;
    const int l16  = lane & 15;
    const int qi   = blockIdx.x;
    const int h    = blockIdx.y;
    const int b    = blockIdx.z;

    const size_t  base = (size_t)(b * NHEADS + h) * SEQ * HDIM;
    const unsigned* Q32 = (const unsigned*)(Qh + base);
    const __bf16* Kbase = Kh + base;
    const __bf16* Vbase = Vt + base;    // [HDIM][SEQ] for this (b,h)

    const unsigned ktBase = (unsigned)(uintptr_t)&Kt [0][0][0];
    const unsigned vtBase = (unsigned)(uintptr_t)&VtT[0][0][0];
    const unsigned KVSZ   = 64 * 72 * 2;  // 9216 B per buffer

    // Q A-fragments held in registers (2 K-chunks over HDIM=64)
    BF16Frag aQ[2];
    {
        int m = 64 * qi + 16 * w + l16;
#pragma unroll
        for (int c = 0; c < 2; ++c)
#pragma unroll
            for (int i = 0; i < 8; ++i) {
                int d = 32 * c + ((i < 4) ? (2 * i + 8 * hl) : (16 + 2 * (i - 4) + 8 * hl));
                aQ[c].u[i] = Q32[(m * HDIM + d) >> 1];
            }
    }

    F8 O[4];
    float mrow[8], lrow[8];
#pragma unroll
    for (int nt = 0; nt < 4; ++nt)
#pragma unroll
        for (int r = 0; r < 8; ++r) O[nt].f[r] = 0.0f;
#pragma unroll
    for (int r = 0; r < 8; ++r) { mrow[r] = -3.0e38f; lrow[r] = 0.0f; }

    // issue one 64x64 K tile + 64x64 V tile (8 async b128 ops per thread)
#define KV_ISSUE(j, buf)                                                           \
    {                                                                              \
        _Pragma("unroll")                                                          \
        for (int i = 0; i < 4; ++i) {                                              \
            int q = tid + 128 * i; int t = q >> 3, c = q & 7;                      \
            async_b128(ktBase + (buf) * KVSZ + t * 144 + c * 16,                   \
                       Kbase + (size_t)(64 * (j) + t) * HDIM + 8 * c);             \
        }                                                                          \
        _Pragma("unroll")                                                          \
        for (int i = 0; i < 4; ++i) {                                              \
            int q = tid + 128 * i; int d = q >> 3, c = q & 7;                      \
            async_b128(vtBase + (buf) * KVSZ + d * 144 + c * 16,                   \
                       Vbase + (size_t)d * SEQ + 64 * (j) + 8 * c);                \
        }                                                                          \
    }

    KV_ISSUE(0, 0);

    for (int j = 0; j <= qi; ++j) {
        const int buf = j & 1;
        if (j < qi) {
            KV_ISSUE(j + 1, buf ^ 1);
            asm volatile("s_wait_asynccnt 0x8" ::: "memory");
        } else {
            asm volatile("s_wait_asynccnt 0x0" ::: "memory");
        }
        __syncthreads();

        // S = Q @ K^T
        F8 sc[4];
#pragma unroll
        for (int nt = 0; nt < 4; ++nt) {
#pragma unroll
            for (int r = 0; r < 8; ++r) sc[nt].f[r] = 0.0f;
#pragma unroll
            for (int c = 0; c < 2; ++c) {
                BF16Frag bk;
#pragma unroll
                for (int i = 0; i < 8; ++i) {
                    int d = 32 * c + ((i < 4) ? (2 * i + 8 * hl) : (16 + 2 * (i - 4) + 8 * hl));
                    bk.u[i] = *(const unsigned*)&Kt[buf][nt * 16 + l16][d];
                }
                sc[nt].v = __builtin_amdgcn_wmma_f32_16x16x32_bf16(
                    false, aQ[c].v, false, bk.v, (short)0, sc[nt].v, false, false);
            }
        }

        // scale + causal mask
#pragma unroll
        for (int nt = 0; nt < 4; ++nt)
#pragma unroll
            for (int r = 0; r < 8; ++r) {
                int sG = 64 * qi + 16 * w + r + 8 * hl;
                int tG = 64 * j + nt * 16 + l16;
                float v = sc[nt].f[r] * 0.125f;        // 1/sqrt(64)
                sc[nt].f[r] = (tG <= sG) ? v : -3.0e38f;
            }

        // online softmax (row stats replicated across the 16-lane half)
#pragma unroll
        for (int r = 0; r < 8; ++r) {
            float mloc = fmaxf(fmaxf(sc[0].f[r], sc[1].f[r]),
                               fmaxf(sc[2].f[r], sc[3].f[r]));
            mloc = red_max16(mloc);
            float mnew = fmaxf(mrow[r], mloc);
            float corr = __expf(mrow[r] - mnew);
            float psum = 0.0f;
#pragma unroll
            for (int nt = 0; nt < 4; ++nt) {
                float p = __expf(sc[nt].f[r] - mnew);
                sc[nt].f[r] = p;
                psum += p;
            }
            psum    = red_sum16(psum);
            lrow[r] = lrow[r] * corr + psum;
            mrow[r] = mnew;
#pragma unroll
            for (int nt = 0; nt < 4; ++nt) O[nt].f[r] *= corr;
        }

        // stage P into LDS (rows 16w..16w+15 are wave-private): C-layout -> A-layout
#pragma unroll
        for (int nt = 0; nt < 4; ++nt)
#pragma unroll
            for (int r = 0; r < 8; ++r)
                Pt[16 * w + r + 8 * hl][nt * 16 + l16] = (__bf16)sc[nt].f[r];
        asm volatile("s_wait_dscnt 0x0" ::: "memory");   // same-wave DS ordering

        // O += P @ V
        BF16Frag aP[2];
        {
            int m = 16 * w + l16;
#pragma unroll
            for (int c = 0; c < 2; ++c)
#pragma unroll
                for (int i = 0; i < 8; ++i) {
                    int t = 32 * c + ((i < 4) ? (2 * i + 8 * hl) : (16 + 2 * (i - 4) + 8 * hl));
                    aP[c].u[i] = *(const unsigned*)&Pt[m][t];
                }
        }
#pragma unroll
        for (int nt = 0; nt < 4; ++nt)
#pragma unroll
            for (int c = 0; c < 2; ++c) {
                BF16Frag bv;
#pragma unroll
                for (int i = 0; i < 8; ++i) {
                    int t = 32 * c + ((i < 4) ? (2 * i + 8 * hl) : (16 + 2 * (i - 4) + 8 * hl));
                    bv.u[i] = *(const unsigned*)&VtT[buf][nt * 16 + l16][t];
                }
                O[nt].v = __builtin_amdgcn_wmma_f32_16x16x32_bf16(
                    false, aP[c].v, false, bv.v, (short)0, O[nt].v, false, false);
            }
        __syncthreads();   // everyone done with Kt/VtT[buf] before it is re-filled
    }
#undef KV_ISSUE

    // epilogue: normalize and write attn output [B,S,DMODEL] bf16
#pragma unroll
    for (int nt = 0; nt < 4; ++nt)
#pragma unroll
        for (int r = 0; r < 8; ++r) {
            int s   = 64 * qi + 16 * w + r + 8 * hl;
            int col = h * HDIM + nt * 16 + l16;
            float v = O[nt].f[r] / lrow[r];
            AO[((size_t)(b * SEQ + s)) * DMODEL + col] = (__bf16)v;
        }
}

// ---------------------------------------------------------------- launch
extern "C" void kernel_launch(void* const* d_in, const int* in_sizes, int n_in,
                              void* d_out, int out_size, void* d_ws, size_t ws_size,
                              hipStream_t stream) {
    const float* x  = (const float*)d_in[0];
    const float* Wq = (const float*)d_in[1];
    const float* bq = (const float*)d_in[2];
    const float* Wk = (const float*)d_in[3];
    const float* bk = (const float*)d_in[4];
    const float* Wv = (const float*)d_in[5];
    const float* bv = (const float*)d_in[6];
    const float* Wo = (const float*)d_in[7];
    const float* bo = (const float*)d_in[8];

    char* ws = (char*)d_ws;
    const size_t MB = 1024u * 1024u;
    __bf16* xb  = (__bf16*)(ws);              // 8 MB
    __bf16* Wqt = (__bf16*)(ws +  8 * MB);    // 2 MB each, TRANSPOSED [N][K]
    __bf16* Wkt = (__bf16*)(ws + 10 * MB);
    __bf16* Wvt = (__bf16*)(ws + 12 * MB);
    __bf16* Wot = (__bf16*)(ws + 14 * MB);
    __bf16* Qh  = (__bf16*)(ws + 16 * MB);    // 8 MB, [B,H,S,HDIM]
    __bf16* Kh  = (__bf16*)(ws + 24 * MB);    // 8 MB, [B,H,S,HDIM]
    __bf16* Vt  = (__bf16*)(ws + 32 * MB);    // 8 MB, [B,H,HDIM,S]
    __bf16* AO  = (__bf16*)(ws + 40 * MB);    // 8 MB, [B,S,DMODEL]

    const int nx = MROWS * DMODEL;            // 4,194,304
    const int nw = DMODEL * DMODEL;           // 1,048,576
    f32_to_bf16_kernel  <<<(nx + 255) / 256, 256, 0, stream>>>(x,  xb, nx);
    f32_to_bf16_T_kernel<<<(nw + 255) / 256, 256, 0, stream>>>(Wq, Wqt);
    f32_to_bf16_T_kernel<<<(nw + 255) / 256, 256, 0, stream>>>(Wk, Wkt);
    f32_to_bf16_T_kernel<<<(nw + 255) / 256, 256, 0, stream>>>(Wv, Wvt);
    f32_to_bf16_T_kernel<<<(nw + 255) / 256, 256, 0, stream>>>(Wo, Wot);

    dim3 gg(DMODEL / 64, MROWS / 64);         // (16, 64)
    gemm_bf16_kernel<<<gg, 128, 0, stream>>>(xb, Wqt, bq, nullptr, Qh, 1);
    gemm_bf16_kernel<<<gg, 128, 0, stream>>>(xb, Wkt, bk, nullptr, Kh, 1);
    gemm_bf16_kernel<<<gg, 128, 0, stream>>>(xb, Wvt, bv, nullptr, Vt, 2);

    dim3 fg(SEQ / 64, NHEADS, BATCH);         // (32, 16, 2)
    flash_attn_kernel<<<fg, 128, 0, stream>>>(Qh, Kh, Vt, AO);

    gemm_bf16_kernel<<<gg, 128, 0, stream>>>(AO, Wot, bo, (float*)d_out, nullptr, 0);
}
